// ConvAConnect_26792005992769
// MI455X (gfx1250) — matrix operand order repaired and checked by reading, not compiled, not run on hardware
//
#include <hip/hip_runtime.h>

typedef __attribute__((ext_vector_type(16))) __bf16 v16bf;
typedef __attribute__((ext_vector_type(8)))  float  v8f;

#define B_    32
#define H_    64
#define W_    64
#define CIN   64
#define FH    3
#define FW    3
#define COUT  128
#define HO    62
#define WO    62
#define NPIX  (HO * WO)          // 3844
#define KTOT  (FH * FW * CIN)    // 576
#define KC    32                 // K per WMMA
#define NKC   (KTOT / KC)        // 18
#define MT_BLK 128               // M rows per workgroup (2 waves x 64)
#define MTILES ((NPIX + MT_BLK - 1) / MT_BLK)  // 31

// ---- helpers -------------------------------------------------------------

__device__ __forceinline__ __bf16 f2bf(float f) {
    union { float f; unsigned u; } in; in.f = f;
    unsigned r = in.u + 0x7FFFu + ((in.u >> 16) & 1u);   // round-to-nearest-even
    union { unsigned short s; __bf16 b; } out;
    out.s = (unsigned short)(r >> 16);
    return out.b;
}

// ---- pass 1: X f32 -> bf16 ----------------------------------------------

__global__ void x_to_bf16(const float* __restrict__ X,
                          __bf16* __restrict__ Xbf, int n4) {
    int i = blockIdx.x * blockDim.x + threadIdx.x;
    if (i >= n4) return;
    float4 v = reinterpret_cast<const float4*>(X)[i];
    union { __bf16 o[4]; uint2 u; } pk;
    pk.o[0] = f2bf(v.x); pk.o[1] = f2bf(v.y);
    pk.o[2] = f2bf(v.z); pk.o[3] = f2bf(v.w);
    reinterpret_cast<uint2*>(Xbf)[i] = pk.u;
}

// ---- pass 2: build per-sample noisy weights, pre-swizzled to B-fragment --
// Layout: Bbuf[b][kc][g][lane][i]  (g = column group of 16, i = 0..15 bf16)
//   value = (W * Werr[loc])[k][col],  k = kc*32 + (i<8 ? 8h+i : 16+8h+i-8), h = lane>=16,
//   col = g*16 + (lane&15).  k ordering == (fh*FW+fw)*CIN + ci == flat HWIO.

__global__ void build_Bmat(const float* __restrict__ W,
                           const float* __restrict__ Werr,
                           const int*   __restrict__ loc_id,
                           __bf16* __restrict__ Bbuf) {
    int t = blockIdx.x * blockDim.x + threadIdx.x;   // 294912 threads, 8 elems each
    if (t >= B_ * NKC * 8 * 32 * 2) return;
    int i0   = (t & 1) * 8;
    int lane = (t >> 1) & 31;
    int g    = (t >> 6) & 7;
    int rest = t >> 9;
    int kc   = rest % NKC;
    int b    = rest / NKC;

    int col  = g * 16 + (lane & 15);
    int loc  = loc_id[b];
    const float* Ep = Werr + (size_t)loc * (KTOT * COUT);

    int kbase = kc * KC + ((lane < 16) ? 0 : 8) + i0 * 2;
    union { __bf16 o[8]; uint4 u; } pk;
#pragma unroll
    for (int j = 0; j < 8; ++j) {
        size_t idx = (size_t)(kbase + j) * COUT + col;
        pk.o[j] = f2bf(W[idx] * Ep[idx]);
    }
    size_t flat = ((((size_t)b * NKC + kc) * 8 + g) * 32 + lane) * 16 + i0;
    *reinterpret_cast<uint4*>(Bbuf + flat) = pk.u;
}

// ---- pass 3: implicit-im2col GEMM via V_WMMA_F32_16X16X32_BF16 -----------
// 8 waves / block arranged 2(M) x 4(N); each wave computes a 64x32 C tile
// (8 accumulators): per k-chunk, 4 A frags + 2 B frags = 12 b128 loads
// feeding 8 WMMAs (1.5 loads/WMMA).

__global__ __launch_bounds__(256)
void conv_wmma(const __bf16* __restrict__ Xbf,
               const __bf16* __restrict__ Bbuf,
               const float*  __restrict__ bias,
               const float*  __restrict__ Berr,
               const int*    __restrict__ loc_id,
               float* __restrict__ out) {
    const int b    = blockIdx.y;
    const int m0   = blockIdx.x * MT_BLK;
    const int w    = threadIdx.x >> 5;
    const int wm   = w >> 2;               // 0..1 : M sub-block of 64
    const int wn   = w & 3;                // 0..3 : N sub-block of 32
    const int lane = threadIdx.x & 31;
    const int mrow = lane & 15;
    const bool hi  = lane >= 16;
    const int lo   = hi ? 8 : 0;           // A K-offset for upper half-wave

    // pixel coords for the wave's four M sub-tiles (clamped: EXEC stays all-1s)
    int oy[4], ox[4];
#pragma unroll
    for (int s = 0; s < 4; ++s) {
        int p = m0 + wm * 64 + s * 16 + mrow;
        if (p > NPIX - 1) p = NPIX - 1;
        oy[s] = p / WO;
        ox[s] = p - oy[s] * WO;
    }

    const uint4* Xq = reinterpret_cast<const uint4*>(Xbf);
    const uint4* Bq = reinterpret_cast<const uint4*>(Bbuf);

    v8f acc[4][2] = {};
#pragma unroll
    for (int kc = 0; kc < NKC; ++kc) {
        const int cell = kc >> 1;           // (fh,fw) cell: 64 k-values = 2 chunks
        const int fh   = cell / 3;
        const int fw   = cell - fh * 3;
        const int cib  = (kc & 1) << 5;     // ci base: 0 or 32

        // A fragments (16x32 bf16): two contiguous 8-elem runs per lane
        union { v16bf v; uint4 u[2]; } A[4];
#pragma unroll
        for (int s = 0; s < 4; ++s) {
            size_t e = (((size_t)b * H_ + (oy[s] + fh)) * W_ + (ox[s] + fw)) * CIN
                       + cib + lo;
            A[s].u[0] = Xq[e >> 3];         // K = cib + lo + {0..7}
            A[s].u[1] = Xq[(e + 16) >> 3];  // K = cib + 16 + lo + {0..7}
        }

        // B fragments: pre-swizzled, 32 contiguous bytes per lane; col-groups
        // g0 = wn*2, g1 = wn*2+1; group stride = 32 lanes * 2 uint4
        size_t be = ((((size_t)b * NKC + kc) * 8 + wn * 2) * 32 + lane) * 2;
        union { v16bf v; uint4 u[2]; } B0, B1;
        B0.u[0] = Bq[be];       B0.u[1] = Bq[be + 1];
        B1.u[0] = Bq[be + 64];  B1.u[1] = Bq[be + 65];

#pragma unroll
        for (int s = 0; s < 4; ++s) {
            acc[s][0] = __builtin_amdgcn_wmma_f32_16x16x32_bf16(
                false, A[s].v, false, B0.v, (short)0, acc[s][0], false, false);
            acc[s][1] = __builtin_amdgcn_wmma_f32_16x16x32_bf16(
                false, A[s].v, false, B1.v, (short)0, acc[s][1], false, false);
        }
    }

    // Epilogue: + bias*Berr[loc]; C layout: VGPR r -> M = r (lanes 0-15) / r+8
    const int loc  = loc_id[b];
    const int col0 = wn * 32 + mrow;
    const int col1 = col0 + 16;
    const float mb0 = bias[col0] * Berr[(size_t)loc * COUT + col0];
    const float mb1 = bias[col1] * Berr[(size_t)loc * COUT + col1];
    const int rb = hi ? 8 : 0;
#pragma unroll
    for (int s = 0; s < 4; ++s) {
#pragma unroll
        for (int r = 0; r < 8; ++r) {
            int p = m0 + wm * 64 + s * 16 + rb + r;
            if (p < NPIX) {
                size_t o = ((size_t)b * NPIX + p) * COUT;
                out[o + col0] = acc[s][0][r] + mb0;
                out[o + col1] = acc[s][1][r] + mb1;
            }
        }
    }
}

// ---- launch --------------------------------------------------------------

extern "C" void kernel_launch(void* const* d_in, const int* in_sizes, int n_in,
                              void* d_out, int out_size, void* d_ws, size_t ws_size,
                              hipStream_t stream) {
    const float* X    = (const float*)d_in[0];
    const float* W    = (const float*)d_in[1];
    const float* bias = (const float*)d_in[2];
    const float* Werr = (const float*)d_in[3];
    const float* Berr = (const float*)d_in[4];
    const int*   loc  = (const int*)d_in[5];
    float* out = (float*)d_out;

    // workspace: Xbf (16 MiB) | Bbuf (4.5 MiB)
    __bf16* Xbf  = (__bf16*)d_ws;
    __bf16* Bbuf = (__bf16*)((char*)d_ws + (size_t)B_ * H_ * W_ * CIN * 2);

    // pass 1: 8.39M f32 -> bf16, 4 per thread
    {
        int n4 = (B_ * H_ * W_ * CIN) / 4;           // 2,097,152
        x_to_bf16<<<(n4 + 255) / 256, 256, 0, stream>>>(X, Xbf, n4);
    }
    // pass 2: 2.36M weight elems, 8 per thread
    {
        int nt = B_ * NKC * 8 * 32 * 2;              // 294,912
        build_Bmat<<<(nt + 255) / 256, 256, 0, stream>>>(W, Werr, loc, Bbuf);
    }
    // pass 3: GEMM, grid = (M tiles, samples)
    {
        dim3 grid(MTILES, B_);
        conv_wmma<<<grid, 256, 0, stream>>>(Xbf, Bbuf, bias, Berr, loc, out);
    }
}
